// FlashMultiHeadSelfAttention_10316511445117
// MI455X (gfx1250) — matrix operand order repaired and checked by reading, not compiled, not run on hardware
//
#include <hip/hip_runtime.h>
#include <hip/hip_bf16.h>
#include <cstdint>

// MI455X / gfx1250: wave32, bf16 WMMA 16x16x32 with f32 accumulation,
// async Global->LDS (ASYNCcnt) double-buffered GEMM staging.

typedef __attribute__((ext_vector_type(16))) __bf16 v16bf;
typedef __attribute__((ext_vector_type(8)))  __bf16 v8bf;
typedef __attribute__((ext_vector_type(8)))  float  v8f;

union V16 { v16bf v; v8bf h[2]; __bf16 e[16]; };

static __device__ inline v8f v8f_zero() {
    v8f z;
#pragma unroll
    for (int i = 0; i < 8; ++i) z[i] = 0.0f;
    return z;
}

#define WMMA_BF16(a, b, c) \
    __builtin_amdgcn_wmma_f32_16x16x32_bf16(false, (a), false, (b), (short)0, (c), false, false)

// Async 16-byte Global->LDS copy (per lane). VDST = LDS byte address,
// VADDR = 64-bit global address, saddr = off. Tracked by ASYNCcnt.
static __device__ inline void async_copy16(unsigned int lds_addr, const float* g) {
    asm volatile("global_load_async_to_lds_b128 %0, %1, off"
                 :: "v"(lds_addr), "v"((unsigned long long)(uintptr_t)g)
                 : "memory");
}
static __device__ inline void wait_async0() {
    asm volatile("s_wait_asynccnt 0" ::: "memory");
}

// ---------------------------------------------------------------------------
// C = A(MxK,f32,row) * B(KxN,f32,row) -> C(MxN,f32), bf16 WMMA inside.
// Block tile 128x256, K-step 32. Tiles staged as f32 via async Global->LDS
// with ping-pong buffers; fp32->bf16 conversion happens at fragment build.
// 8 waves in a 2x4 grid; each wave computes a 64x64 subtile = 4x4 16x16
// fragments -> 16 WMMAs per K-step per wave.
// ---------------------------------------------------------------------------
__global__ __launch_bounds__(256)
void gemm_f32_bf16wmma(const float* __restrict__ A, const float* __restrict__ Bm,
                       float* __restrict__ C, int M, int N, int K)
{
    __shared__ float Af[2][128][32];   // 2 x 16 KB, row-major [m][k]
    __shared__ float Bf[2][32][256];   // 2 x 32 KB, row-major [k][n]

    const int tid  = threadIdx.x;
    const int lane = tid & 31;
    const int wid  = tid >> 5;
    const int wm   = wid >> 2;        // 0..1 -> M offset wm*64
    const int wn   = wid & 3;         // 0..3 -> N offset wn*64
    const int bm   = blockIdx.y * 128;
    const int bn   = blockIdx.x * 256;
    const int lr   = lane & 15;
    const int hb   = lane >> 4;
    const int kbA  = hb * 8;          // A-fragment half-K base
    const int kbB  = hb * 16;         // unused; B fragments also use kbA

    v8f acc[4][4];
#pragma unroll
    for (int mi = 0; mi < 4; ++mi)
#pragma unroll
        for (int ni = 0; ni < 4; ++ni) acc[mi][ni] = v8f_zero();

    // Issue one 128x32 A tile (4 ops) + 32x256 B tile (8 ops) per thread.
    auto issue_tile = [&](int kt, int buf) {
#pragma unroll
        for (int q = 0; q < 4; ++q) {           // A: 32 rows per issue
            const int r = q * 32 + (tid >> 3);
            const int c = (tid & 7) * 4;
            async_copy16((unsigned int)(uintptr_t)&Af[buf][r][c],
                         A + (size_t)(bm + r) * K + kt + c);
        }
#pragma unroll
        for (int q = 0; q < 8; ++q) {           // B: 4 k-rows per issue
            const int k = q * 4 + (tid >> 6);
            const int n = (tid & 63) * 4;
            async_copy16((unsigned int)(uintptr_t)&Bf[buf][k][n],
                         Bm + (size_t)(kt + k) * N + bn + n);
        }
    };

    issue_tile(0, 0);
    const int nK = K >> 5;
    for (int it = 0; it < nK; ++it) {
        const int buf = it & 1;
        wait_async0();         // this wave's tile slices have landed in LDS
        __syncthreads();       // everyone's slices landed; prior reads finished
        if (it + 1 < nK) issue_tile((it + 1) << 5, buf ^ 1);

        // A fragments: row = lane&15; per-lane K halves [kb,kb+8) u [kb+16,kb+24)
        V16 af[4];
#pragma unroll
        for (int mi = 0; mi < 4; ++mi) {
            const float* p = &Af[buf][wm * 64 + mi * 16 + lr][0];
            const float4 c0 = *(const float4*)(p + kbA);
            const float4 c1 = *(const float4*)(p + kbA + 4);
            const float4 c2 = *(const float4*)(p + kbA + 16);
            const float4 c3 = *(const float4*)(p + kbA + 20);
            af[mi].e[0]  = (__bf16)c0.x; af[mi].e[1]  = (__bf16)c0.y;
            af[mi].e[2]  = (__bf16)c0.z; af[mi].e[3]  = (__bf16)c0.w;
            af[mi].e[4]  = (__bf16)c1.x; af[mi].e[5]  = (__bf16)c1.y;
            af[mi].e[6]  = (__bf16)c1.z; af[mi].e[7]  = (__bf16)c1.w;
            af[mi].e[8]  = (__bf16)c2.x; af[mi].e[9]  = (__bf16)c2.y;
            af[mi].e[10] = (__bf16)c2.z; af[mi].e[11] = (__bf16)c2.w;
            af[mi].e[12] = (__bf16)c3.x; af[mi].e[13] = (__bf16)c3.y;
            af[mi].e[14] = (__bf16)c3.z; af[mi].e[15] = (__bf16)c3.w;
        }
        // B fragments: col = lane&15, K halves [kb,kb+8) u [kb+16,kb+24) per lane -> strided [k][n]
        V16 bv[4];
#pragma unroll
        for (int ni = 0; ni < 4; ++ni) {
            const int n = wn * 64 + ni * 16 + lr;
#pragma unroll
            for (int i = 0; i < 16; ++i)
                bv[ni].e[i] = (__bf16)Bf[buf][kbA + i + (i >= 8 ? 8 : 0)][n];
        }
#pragma unroll
        for (int mi = 0; mi < 4; ++mi)
#pragma unroll
            for (int ni = 0; ni < 4; ++ni)
                acc[mi][ni] = WMMA_BF16(af[mi].v, bv[ni].v, acc[mi][ni]);
    }

    // C/D layout: VGPR r, lane l -> row = r + (l>>4)*8, col = l&15.
#pragma unroll
    for (int mi = 0; mi < 4; ++mi)
#pragma unroll
        for (int ni = 0; ni < 4; ++ni) {
            const int col  = bn + wn * 64 + ni * 16 + lr;
            const int row0 = bm + wm * 64 + mi * 16 + hb * 8;
#pragma unroll
            for (int r = 0; r < 8; ++r)
                C[(size_t)(row0 + r) * N + col] = acc[mi][ni][r];
        }
}

// ---------------------------------------------------------------------------
// RoPE on Q,K + repack: qkv[b,s,3,h,d] f32 ->
//   Qb,Kb bf16 [B,H,S,D]   (rows = contiguous head-dim -> B-fragments for QK^T)
//   Vt    bf16 [B,H,D,S]   (rows = contiguous seq     -> B-fragments for P*V)
// ---------------------------------------------------------------------------
__global__ __launch_bounds__(256)
void rope_pack(const float* __restrict__ qkv, __bf16* __restrict__ Qb,
               __bf16* __restrict__ Kb, __bf16* __restrict__ Vt)
{
    const int gid = blockIdx.x * 256 + threadIdx.x;   // B*S*H*32 threads
    const int j = gid & 31;            // rotary pair index (0..31)
    const int h = (gid >> 5) & 15;
    const int s = (gid >> 9) & 2047;
    const int b = gid >> 20;

    const size_t base = ((size_t)(b * 2048 + s)) * 3072 + h * 64 + j;
    const float q1 = qkv[base];
    const float q2 = qkv[base + 32];
    const float k1 = qkv[base + 1024];
    const float k2 = qkv[base + 1024 + 32];
    const float v1 = qkv[base + 2048];
    const float v2 = qkv[base + 2048 + 32];

    const float inv = __powf(10000.0f, -(float)j * (1.0f / 32.0f));
    const float ang = (float)s * inv;
    const float c  = __cosf(ang);
    const float sn = __sinf(ang);

    const size_t qo = (((size_t)b * 16 + h) * 2048 + s) * 64;
    Qb[qo + j]      = (__bf16)(q1 * c - q2 * sn);
    Qb[qo + j + 32] = (__bf16)(q2 * c + q1 * sn);
    Kb[qo + j]      = (__bf16)(k1 * c - k2 * sn);
    Kb[qo + j + 32] = (__bf16)(k2 * c + k1 * sn);

    const size_t vo = ((size_t)b * 16 + h) * 64;
    Vt[(vo + j) * 2048 + s]      = (__bf16)v1;
    Vt[(vo + j + 32) * 2048 + s] = (__bf16)v2;
}

// ---------------------------------------------------------------------------
// Flash attention, one (b,h) per block.y/z, 128 q-rows per block, 16 per wave.
// Online softmax over S=2048 in K-tiles of 64. Score frags (C-layout) are
// reduced with 16-lane shfl_xor, exp'd, bounced through per-wave LDS to
// re-layout as A fragments, then P*V accumulates via WMMA into f32 frags.
// ---------------------------------------------------------------------------
__global__ __launch_bounds__(256)
void flash_attn_wmma(const __bf16* __restrict__ Qb, const __bf16* __restrict__ Kb,
                     const __bf16* __restrict__ Vt, float* __restrict__ attn)
{
    __shared__ __bf16 plds[8][16][64];   // per-wave P tile, 16 KB total

    const int tid  = threadIdx.x;
    const int lane = tid & 31;
    const int wid  = tid >> 5;
    const int lr   = lane & 15;
    const int hb   = lane >> 4;
    const int kbA  = hb * 8;
    const int kbB  = hb * 16;

    const int h  = blockIdx.y;
    const int b  = blockIdx.z;
    const int bh = b * 16 + h;
    const int q0 = blockIdx.x * 128 + wid * 16;

    // Q fragments: 16 rows x 64 dims = two 16x32 A-frags.
    V16 aq[2];
    {
        const __bf16* qp = Qb + ((size_t)bh * 2048 + q0 + lr) * 64;
#pragma unroll
        for (int dc = 0; dc < 2; ++dc) {
            aq[dc].h[0] = *(const v8bf*)(qp + dc * 32 + kbA);
            aq[dc].h[1] = *(const v8bf*)(qp + dc * 32 + kbA + 16);
        }
    }

    float m[8], l[8];
#pragma unroll
    for (int r = 0; r < 8; ++r) { m[r] = -1e30f; l[r] = 0.0f; }
    v8f o[4];
#pragma unroll
    for (int df = 0; df < 4; ++df) o[df] = v8f_zero();

    for (int kt = 0; kt < 2048; kt += 64) {
        // prefetch next K/V tiles into cache (global_prefetch_b8)
        if (kt + 64 < 2048) {
            __builtin_prefetch(Kb + ((size_t)bh * 2048 + kt + 64 + lr) * 64, 0, 1);
            __builtin_prefetch(Vt + ((size_t)bh * 64 + lr) * 2048 + kt + 64, 0, 1);
        }

        // ---- scores S = Q K^T (four 16x16 frags, K-dim = 64 = 2 WMMAs each)
        v8f sc[4];
#pragma unroll
        for (int nf = 0; nf < 4; ++nf) sc[nf] = v8f_zero();
#pragma unroll
        for (int nf = 0; nf < 4; ++nf) {
            const int kcol = kt + nf * 16 + lr;
            const __bf16* kp = Kb + ((size_t)bh * 2048 + kcol) * 64;
#pragma unroll
            for (int dc = 0; dc < 2; ++dc) {
                V16 bk;
                bk.h[0] = *(const v8bf*)(kp + dc * 32 + kbA);
                bk.h[1] = *(const v8bf*)(kp + dc * 32 + kbA + 16);
                sc[nf] = WMMA_BF16(aq[dc].v, bk.v, sc[nf]);
            }
        }
#pragma unroll
        for (int nf = 0; nf < 4; ++nf)
#pragma unroll
            for (int r = 0; r < 8; ++r) sc[nf][r] *= 0.125f;   // 1/sqrt(64)

        // ---- online softmax: row max/sum via 16-lane xor reductions
        float mnew[8], fac[8];
#pragma unroll
        for (int r = 0; r < 8; ++r) {
            float t = sc[0][r];
#pragma unroll
            for (int nf = 1; nf < 4; ++nf) t = fmaxf(t, sc[nf][r]);
#pragma unroll
            for (int off = 1; off < 16; off <<= 1)
                t = fmaxf(t, __shfl_xor(t, off, 32));
            mnew[r] = fmaxf(m[r], t);
            fac[r]  = __expf(m[r] - mnew[r]);
            m[r]    = mnew[r];
        }
#pragma unroll
        for (int nf = 0; nf < 4; ++nf)
#pragma unroll
            for (int r = 0; r < 8; ++r)
                sc[nf][r] = __expf(sc[nf][r] - mnew[r]);
#pragma unroll
        for (int r = 0; r < 8; ++r) {
            float t = sc[0][r] + sc[1][r] + sc[2][r] + sc[3][r];
#pragma unroll
            for (int off = 1; off < 16; off <<= 1)
                t += __shfl_xor(t, off, 32);
            l[r] = l[r] * fac[r] + t;
        }
#pragma unroll
        for (int df = 0; df < 4; ++df)
#pragma unroll
            for (int r = 0; r < 8; ++r) o[df][r] *= fac[r];

        // ---- P (C-layout) -> LDS -> A-fragment layout (same wave, in-order DS)
#pragma unroll
        for (int nf = 0; nf < 4; ++nf)
#pragma unroll
            for (int r = 0; r < 8; ++r)
                plds[wid][hb * 8 + r][nf * 16 + lr] = (__bf16)sc[nf][r];

        asm volatile("s_wait_dscnt 0" ::: "memory");

        V16 ap[2];
#pragma unroll
        for (int kc = 0; kc < 2; ++kc) {
            const __bf16* pp = &plds[wid][lr][kc * 32];
            ap[kc].h[0] = *(const v8bf*)(pp + kbA);
            ap[kc].h[1] = *(const v8bf*)(pp + kbA + 16);
        }

        // ---- O += P * V   (V transposed: rows = head-dim, contiguous seq)
#pragma unroll
        for (int df = 0; df < 4; ++df) {
            const __bf16* vp = Vt + ((size_t)bh * 64 + df * 16 + lr) * 2048 + kt;
#pragma unroll
            for (int kc = 0; kc < 2; ++kc) {
                V16 bvv;
                bvv.h[0] = *(const v8bf*)(vp + kc * 32 + kbA);
                bvv.h[1] = *(const v8bf*)(vp + kc * 32 + kbA + 16);
                o[df] = WMMA_BF16(ap[kc].v, bvv.v, o[df]);
            }
        }
    }

    // ---- normalize and write attn[b,s,h,d] f32
#pragma unroll
    for (int df = 0; df < 4; ++df)
#pragma unroll
        for (int r = 0; r < 8; ++r) {
            const int spos = q0 + hb * 8 + r;
            const float val = o[df][r] / l[r];
            attn[(((size_t)b * 2048 + spos) * 16 + h) * 64 + df * 16 + lr] = val;
        }
}

// ---------------------------------------------------------------------------
extern "C" void kernel_launch(void* const* d_in, const int* in_sizes, int n_in,
                              void* d_out, int out_size, void* d_ws, size_t ws_size,
                              hipStream_t stream)
{
    (void)in_sizes; (void)n_in; (void)out_size; (void)ws_size;
    const float* x    = (const float*)d_in[0];   // [2,2048,1024]
    const float* Wqkv = (const float*)d_in[1];   // [1024,3072]
    const float* Wout = (const float*)d_in[2];   // [1024,1024]
    float* out = (float*)d_out;                  // [2,2048,1024]

    char* ws = (char*)d_ws;
    float*  qkv = (float*)ws;                                // 4096x3072 f32 (48 MB)
    const size_t qkv_bytes = (size_t)4096 * 3072 * sizeof(float);
    __bf16* Qb = (__bf16*)(ws + qkv_bytes);                  // [B,H,S,D] bf16 (8 MB)
    __bf16* Kb = Qb + (size_t)2 * 16 * 2048 * 64;            // [B,H,S,D] bf16 (8 MB)
    __bf16* Vt = Kb + (size_t)2 * 16 * 2048 * 64;            // [B,H,D,S] bf16 (8 MB)
    float*  attn = qkv;                                      // reuse qkv region (dead)

    // 1) QKV projection: [4096x1024] x [1024x3072], block tile 128x256
    gemm_f32_bf16wmma<<<dim3(3072 / 256, 4096 / 128), 256, 0, stream>>>(
        x, Wqkv, qkv, 4096, 3072, 1024);
    // 2) RoPE + bf16 repack (V transposed)
    rope_pack<<<(2 * 2048 * 16 * 32) / 256, 256, 0, stream>>>(qkv, Qb, Kb, Vt);
    // 3) Flash attention, 32 (b,h) pairs x 16 q-tiles
    flash_attn_wmma<<<dim3(2048 / 128, 16, 2), 256, 0, stream>>>(Qb, Kb, Vt, attn);
    // 4) Output projection: [4096x1024] x [1024x1024]
    gemm_f32_bf16wmma<<<dim3(1024 / 256, 4096 / 128), 256, 0, stream>>>(
        attn, Wout, out, 4096, 1024, 1024);
}
